// Mex_31447750542208
// MI455X (gfx1250) — matrix operand order repaired and checked by reading, not compiled, not run on hardware
//
#include <hip/hip_runtime.h>

typedef __attribute__((ext_vector_type(16))) _Float16 v16h;
typedef __attribute__((ext_vector_type(8)))  float    v8f;

#define MEX_C   32
#define MEX_H   128
#define MEX_W   128
#define MEX_K   288     // C * 3 * 3
#define PADW    130     // W + 2 (zero pad cols)
#define NINST   32
#define BFRAG_ELEMS (9 * 2 * 32 * 16)   // tap * group * lane * v16h elems

#define L2E 1.44269504088896340736f     // log2(e)
#define LN2 0.69314718055994530942f

// ---------------------------------------------------------------------------
// Kernel 1: per-instance max mo[i] and B = exp(o - mo) packed into the CDNA5
// wave32 WMMA B-fragment layout with the custom K permutation:
//   K-step kk = 3x3 tap r (0..8), K-local = channel c (0..31)
//   element e = ((kk*2 + g)*32 + lane)*16 + jj
//   c = jj + (lane<16 ? 0 : 16), global k = c*9 + kk, inst = g*16 + lane%16
// ---------------------------------------------------------------------------
__global__ __launch_bounds__(256) void mex_prep(const float* __restrict__ off,
                                                _Float16* __restrict__ Bfrag,
                                                float* __restrict__ mo_out) {
  __shared__ float so[NINST * MEX_K];
  __shared__ float smo[NINST];
  const int tid = threadIdx.x;
  for (int e = tid; e < NINST * MEX_K; e += 256) so[e] = off[e];
  __syncthreads();
  if (tid < NINST) {
    float m = so[tid * MEX_K];
    for (int k = 1; k < MEX_K; ++k) m = fmaxf(m, so[tid * MEX_K + k]);
    smo[tid]    = m;
    mo_out[tid] = m;
  }
  __syncthreads();
  for (int e = tid; e < BFRAG_ELEMS; e += 256) {
    int jj   = e & 15;
    int lane = (e >> 4) & 31;
    int g    = (e >> 9) & 1;
    int kk   = e >> 10;                       // tap index r
    int inst = g * 16 + (lane & 15);
    int c    = jj + ((lane < 16) ? 0 : 16);   // channel
    int k    = c * 9 + kk;                    // offsets are (c, kh, kw) order
    Bfrag[e] = (_Float16)__expf(so[inst * MEX_K + k] - smo[inst]);
  }
}

// ---------------------------------------------------------------------------
// Kernel 2: one block per (n, h). 8 waves; each wave owns 16 output columns
// x 32 instances (two 16x16 WMMA accumulators sharing one A fragment).
// ---------------------------------------------------------------------------
__global__ __launch_bounds__(256) void mex_main(const float* __restrict__ x,
                                                const _Float16* __restrict__ Bfrag,
                                                const float* __restrict__ mo,
                                                float* __restrict__ out) {
  // Overlaid shared memory:
  //  [0,12480)        xs : xs[c*390 + dy*130 + col], zero-padded cols/rows
  //  [12480,12870)    m2 : channel-max per (dy,col)
  //  [12870,12998)    mxa: 3x3 window max per output col
  //  after GEMM phase [0,4160) reused as sout[32][130] for coalesced stores
  __shared__ float smem[13000];
  float* xs  = smem;
  float* m2  = smem + 12480;
  float* mxa = smem + 12870;

  const int tid = threadIdx.x;
  const int n   = blockIdx.x >> 7;
  const int h   = blockIdx.x & 127;

  // ---- stage x rows h-1..h+1 into LDS (zero padded rows/cols) ----
  if (tid < MEX_C * 3) {                 // zero pad columns of each (c,dy) row
    xs[tid * PADW]            = 0.f;
    xs[tid * PADW + PADW - 1] = 0.f;
  }
  for (int e = tid; e < MEX_C * 3 * (MEX_W >> 2); e += 256) {
    int row = e >> 5;                    // c*3 + dy
    int q   = e & 31;                    // float4 index along w
    int c   = row / 3;
    int dy  = row - c * 3;
    int hr  = h - 1 + dy;
    float4 v = make_float4(0.f, 0.f, 0.f, 0.f);
    if ((unsigned)hr < (unsigned)MEX_H)
      v = *(const float4*)(x + (((n * MEX_C + c) * MEX_H + hr) * MEX_W + (q << 2)));
    float* dst = xs + row * PADW + 1 + (q << 2);
    dst[0] = v.x; dst[1] = v.y; dst[2] = v.z; dst[3] = v.w;
  }
  __syncthreads();

  // ---- channel max per (dy, col) ----
  for (int e = tid; e < 3 * PADW; e += 256) {
    int dy = e / PADW, col = e - dy * PADW;
    float m = xs[dy * PADW + col];
    for (int c = 1; c < MEX_C; ++c) m = fmaxf(m, xs[(c * 3 + dy) * PADW + col]);
    m2[e] = m;
  }
  __syncthreads();

  // ---- 3x3 window max per output column ----
  if (tid < MEX_W) {
    float m = m2[tid];
#pragma unroll
    for (int dy = 0; dy < 3; ++dy)
#pragma unroll
      for (int dx = 0; dx < 3; ++dx)
        m = fmaxf(m, m2[dy * PADW + tid + dx]);
    mxa[tid] = m;
  }
  __syncthreads();

  // ---- WMMA patch-GEMM: s = exp(p - mx) @ exp(o - mo)^T ----
  const int wave = tid >> 5;
  const int lane = tid & 31;
  const int hi   = lane >> 4;           // which 16-lane half
  const int mrow = lane & 15;           // A-matrix row (both halves hold M=0..15)
  const int w0   = wave * 16;
  const int w    = w0 + mrow;
  const float mxv = mxa[w];
  const float mml = -mxv * L2E;         // folded into exp2 argument

  // Single lane-dependent LDS base; all per-element offsets are immediates.
  const float* base = xs + w + hi * (16 * 3 * PADW);

  v8f acc0 = {};
  v8f acc1 = {};
#pragma unroll
  for (int kk = 0; kk < 9; ++kk) {      // kk = 3x3 tap; dy,dx compile-time
    const int dy = kk / 3;
    const int dx = kk - dy * 3;
    const float* bp = base + dy * PADW + dx;
    v16h a;
#pragma unroll
    for (int jj = 0; jj < 16; ++jj) {
      // A 16-bit layout: channel c = jj + hi*16; hi folded in base
      float xv = bp[jj * (3 * PADW)];
      a[jj] = (_Float16)__builtin_amdgcn_exp2f(__builtin_fmaf(xv, L2E, mml));
    }
    v16h b0 = *(const v16h*)(Bfrag + ((kk * 2 + 0) * 32 + lane) * 16);
    v16h b1 = *(const v16h*)(Bfrag + ((kk * 2 + 1) * 32 + lane) * 16);
    acc0 = __builtin_amdgcn_wmma_f32_16x16x32_f16(false, a, false, b0,
                                                  (short)0, acc0, false, false);
    acc1 = __builtin_amdgcn_wmma_f32_16x16x32_f16(false, a, false, b1,
                                                  (short)0, acc1, false, false);
  }

  // ---- epilogue: out = mx + mo + log(s) - log(K) ----
  const float lk  = 5.66296048013595f;  // ln(288)
  const float mo0 = mo[lane & 15];
  const float mo1 = mo[16 + (lane & 15)];
  float r0[8], r1[8];
#pragma unroll
  for (int r = 0; r < 8; ++r) {
    // C/D layout: VGPR r -> M = r (lanes 0-15) or 8+r (lanes 16-31); N = lane%16
    int   mr   = r + (hi << 3);
    float base_v = mxa[w0 + mr];
    r0[r] = base_v + mo0 + __builtin_amdgcn_logf(acc0[r]) * LN2 - lk;
    r1[r] = base_v + mo1 + __builtin_amdgcn_logf(acc1[r]) * LN2 - lk;
  }
  __syncthreads();                      // xs dead; reuse smem as sout[32][130]
  float* sout = smem;
#pragma unroll
  for (int r = 0; r < 8; ++r) {
    int mr   = r + (hi << 3);
    int wcol = w0 + mr;
    sout[(lane & 15) * PADW + wcol]        = r0[r];
    sout[(16 + (lane & 15)) * PADW + wcol] = r1[r];
  }
  __syncthreads();
  // coalesced float4 stores: out layout (N, I, H, W)
  for (int e = tid; e < NINST * (MEX_W >> 2); e += 256) {
    int inst = e >> 5;
    int q    = e & 31;
    const float* sp = sout + inst * PADW + (q << 2);
    float4 v = make_float4(sp[0], sp[1], sp[2], sp[3]);
    *(float4*)(out + ((n * NINST + inst) * MEX_H + h) * MEX_W + (q << 2)) = v;
  }
}

// ---------------------------------------------------------------------------
extern "C" void kernel_launch(void* const* d_in, const int* in_sizes, int n_in,
                              void* d_out, int out_size, void* d_ws, size_t ws_size,
                              hipStream_t stream) {
  const float* x   = (const float*)d_in[0];   // (8, 32, 128, 128) f32
  const float* off = (const float*)d_in[1];   // (1, 32, 32, 3, 3) f32
  float* out = (float*)d_out;                 // (8, 32, 128, 128) f32

  _Float16* Bfrag = (_Float16*)d_ws;                              // 18432 B
  float*    mo    = (float*)((char*)d_ws + BFRAG_ELEMS * sizeof(_Float16));

  mex_prep<<<1, 256, 0, stream>>>(off, Bfrag, mo);
  mex_main<<<8 * MEX_H, 256, 0, stream>>>(x, Bfrag, mo, out);
}
